// MicroCortexMLA_77395310674354
// MI455X (gfx1250) — matrix operand order repaired and checked by reading, not compiled, not run on hardware
//
#include <hip/hip_runtime.h>
#include <hip/hip_bf16.h>
#include <math.h>

// ---------------- CDNA5 WMMA plumbing (wave32, 16x16x32 bf16 -> f32) -------
typedef __attribute__((ext_vector_type(16))) __bf16 v16bf;
typedef __attribute__((ext_vector_type(8)))  float  v8f;
typedef __attribute__((__vector_size__(16))) int    v4i;

union Frag16 { v16bf v; unsigned u[8]; };

#define AS1 __attribute__((address_space(1)))
#define AS3 __attribute__((address_space(3)))

#if __has_builtin(__builtin_amdgcn_global_load_async_to_lds_b128)
#define HAVE_ASYNC_LDS 1
#else
#define HAVE_ASYNC_LDS 0
#endif

__device__ __forceinline__ void wait_asynccnt0() {
#if __has_builtin(__builtin_amdgcn_s_wait_asynccnt)
  __builtin_amdgcn_s_wait_asynccnt(0);
#else
  asm volatile("s_wait_asynccnt 0x0" ::: "memory");
#endif
}

__device__ __forceinline__ unsigned short f2bf(float x) {
  unsigned u = __float_as_uint(x);
  u += 0x7FFFu + ((u >> 16) & 1u);           // round-to-nearest-even
  return (unsigned short)(u >> 16);
}

__device__ __forceinline__ unsigned f2bf2(float lo, float hi) {
  return (unsigned)f2bf(lo) | ((unsigned)f2bf(hi) << 16);
}

__device__ __forceinline__ v8f v8f_zero() {
  v8f z = {0.f, 0.f, 0.f, 0.f, 0.f, 0.f, 0.f, 0.f};
  return z;
}

__device__ __forceinline__ v8f wmma_bf16(v16bf a, v16bf b, v8f c) {
  // (neg_a, A, neg_b, B, c_mod, C, reuse_a, reuse_b)
  return __builtin_amdgcn_wmma_f32_16x16x32_bf16(false, a, false, b, (short)0, c,
                                                 false, false);
}

// A fragment 16x32 bf16. Element (m = lane&15, k). base points at (m=0,k=0),
// ld = row stride in elements. Pairs contiguous -> u32 loads (merge to b128).
__device__ __forceinline__ v16bf load_afrag(const unsigned short* base, int ld) {
  int lane = threadIdx.x & 31;
  const unsigned short* p = base + (size_t)(lane & 15) * ld;
  int khalf = (lane >> 4) * 16;
  Frag16 f;
#pragma unroll
  for (int j = 0; j < 8; ++j) {
    int kk = 2 * j + khalf;
    f.u[j] = *(const unsigned*)(p + kk);
  }
  return f.v;
}

// B fragment 32x16 bf16, element (k, n = lane&15) at base + n*ld + k (k contig)
__device__ __forceinline__ v16bf load_bfrag_kcont(const unsigned short* base, int ld) {
  int lane = threadIdx.x & 31;
  const unsigned short* p = base + (size_t)(lane & 15) * ld;
  int khalf = (lane >> 4) * 16;
  Frag16 f;
#pragma unroll
  for (int j = 0; j < 8; ++j)
    f.u[j] = *(const unsigned*)(p + khalf + 2 * j);
  return f.v;
}

// B fragment 32x16 bf16, element (k, n = lane&15) at base + k*ld + n (k strided)
__device__ __forceinline__ v16bf load_bfrag_kstride(const unsigned short* base, int ld) {
  int lane = threadIdx.x & 31;
  int n = lane & 15;
  int khalf = (lane >> 4) * 16;
  Frag16 f;
#pragma unroll
  for (int j = 0; j < 8; ++j) {
    unsigned lo = base[(size_t)(khalf + 2 * j) * ld + n];
    unsigned hi = base[(size_t)(khalf + 2 * j + 1) * ld + n];
    f.u[j] = lo | (hi << 16);
  }
  return f.v;
}

// ---------------- Generic tiled GEMM: C[M,N] = A[M,K] * B^T ----------------
// A row-major f32 (row stride lda). B element (n,k) at Bw[n*bStrideN+k*bStrideK].
// Batched over grid.z: z -> (zo=z/zdiv, zi=z%zdiv). BF16OUT chosen at compile time.
// Tiles: BM=128, BN=64, BK=32; 128 threads = 4 waves in 2x2; wave = 64x32 ->
// 8 WMMAs per 6 fragment loads.
template <bool BF16OUT>
__global__ __launch_bounds__(128) void gemm_nt(
    const float* __restrict__ A, const float* __restrict__ Bw,
    void* __restrict__ Cv,
    int K, int lda, long bStrideN, long bStrideK, int ldc,
    long aOffOuter, long aOffInner, long bOffInner, long cOffZ, int zdiv)
{
  __shared__ __align__(16) unsigned short As[128 * 32];
  __shared__ __align__(16) unsigned short Bs[64 * 32];
  int z = blockIdx.z;
  int zo = z / zdiv, zi = z - zo * zdiv;
  A  += (size_t)zo * aOffOuter + (size_t)zi * aOffInner;
  Bw += (size_t)zi * bOffInner;
  size_t cbase = (size_t)cOffZ * z;
  int m0 = blockIdx.y * 128;
  int n0 = blockIdx.x * 64;
  int tid = threadIdx.x;
  int wave = tid >> 5, lane = tid & 31;
  int wr = (wave >> 1) * 64, wc = (wave & 1) * 32;

  v8f acc[4][2];
#pragma unroll
  for (int im = 0; im < 4; ++im)
#pragma unroll
    for (int in = 0; in < 2; ++in) acc[im][in] = v8f_zero();

  for (int k0 = 0; k0 < K; k0 += 32) {
    __syncthreads();
#pragma unroll
    for (int i = 0; i < 16; ++i) {            // A: 128x32 elems, 2/thread/iter
      int e = (i * 128 + tid) * 2;
      int r = e >> 5, kk = e & 31;
      const float* ap = A + (size_t)(m0 + r) * lda + (k0 + kk);
      *(unsigned*)&As[e] = f2bf2(ap[0], ap[1]);
    }
#pragma unroll
    for (int i = 0; i < 8; ++i) {             // B: 64x32 elems
      int e = (i * 128 + tid) * 2;
      int r = e >> 5, kk = e & 31;
      const float* bp = Bw + (size_t)(n0 + r) * bStrideN + (size_t)(k0 + kk) * bStrideK;
      *(unsigned*)&Bs[e] = f2bf2(bp[0], bp[bStrideK]);
    }
    __syncthreads();
    v16bf bfr[2];
#pragma unroll
    for (int in = 0; in < 2; ++in)
      bfr[in] = load_bfrag_kcont(Bs + (wc + in * 16) * 32, 32);
#pragma unroll
    for (int im = 0; im < 4; ++im) {
      v16bf a = load_afrag(As + (wr + im * 16) * 32, 32);
#pragma unroll
      for (int in = 0; in < 2; ++in)
        acc[im][in] = wmma_bf16(a, bfr[in], acc[im][in]);
    }
  }

  int mb = (lane >> 4) * 8, nn = lane & 15;
#pragma unroll
  for (int im = 0; im < 4; ++im)
#pragma unroll
    for (int in = 0; in < 2; ++in)
#pragma unroll
      for (int j = 0; j < 8; ++j) {
        size_t row = (size_t)m0 + wr + im * 16 + mb + j;
        size_t col = (size_t)n0 + wc + in * 16 + nn;
        if constexpr (BF16OUT)
          ((unsigned short*)Cv)[cbase + row * ldc + col] = f2bf(acc[im][in][j]);
        else
          ((float*)Cv)[cbase + row * ldc + col] = acc[im][in][j];
      }
}

// ---------------- RMSNorm over rows of 1536 --------------------------------
__global__ __launch_bounds__(256) void rmsnorm1536(float* __restrict__ x,
                                                   const float* __restrict__ w) {
  __shared__ float red[256];
  size_t row = blockIdx.x;
  float* p = x + row * 1536;
  int tid = threadIdx.x;
  float v[6];
  float ss = 0.f;
#pragma unroll
  for (int i = 0; i < 6; ++i) { v[i] = p[tid + i * 256]; ss += v[i] * v[i]; }
  red[tid] = ss;
  __syncthreads();
  for (int s = 128; s > 0; s >>= 1) {
    if (tid < s) red[tid] += red[tid + s];
    __syncthreads();
  }
  float inv = rsqrtf(red[0] * (1.0f / 1536.f) + 1e-6f);
#pragma unroll
  for (int i = 0; i < 6; ++i) { int c = tid + i * 256; p[c] = w[c] * (v[i] * inv); }
}

// ---------------- RoPE helpers ---------------------------------------------
__device__ __forceinline__ float rope_elem(const float* x, int i, float pos) {
  int j = i & 31;
  float invf = __expf(-(float)j * (13.815510558f / 32.0f));  // 1e6^(-j/32)
  float ang = pos * invf;
  float c = __cosf(ang), sn = __sinf(ang);
  float x0 = x[2 * j], x1 = x[2 * j + 1];
  return (i < 32) ? (x0 * c - x1 * sn) : (x1 * c + x0 * sn);
}

// kcat[B*S, 576] bf16 : cols 0..63 = rope(k_rope) ; 64..575 = ckv latent
__global__ void build_kcat(const float* __restrict__ ckv,
                           const int* __restrict__ pos_ids,
                           unsigned short* __restrict__ kcat) {
  size_t row = blockIdx.x;
  int i = threadIdx.x;                       // 0..575
  const float* src = ckv + row * 576;
  float out;
  if (i < 64) out = rope_elem(src + 512, i, (float)pos_ids[row]);
  else        out = src[i - 64];
  kcat[row * 576 + i] = f2bf(out);
}

// qcat[(b*16+h)*S+s, 0..63] = rope(q_rope) from q[b,s,h,128..191]
__global__ __launch_bounds__(256) void rope_q(const float* __restrict__ q,
                                              const int* __restrict__ pos_ids,
                                              unsigned short* __restrict__ qcat, int S) {
  size_t idx = (size_t)blockIdx.x * 256 + threadIdx.x;  // B*NH*S*64 total
  int i = (int)(idx & 63);
  size_t r = idx >> 6;                    // z*S + s
  int s = (int)(r % (size_t)S);
  size_t z = r / (size_t)S;
  int h = (int)(z & 15);
  size_t b = z >> 4;
  const float* src = q + (((b * S + s) * 16 + h) * 192 + 128);
  float out = rope_elem(src, i, (float)pos_ids[b * S + s]);
  qcat[r * 576 + i] = f2bf(out);
}

__global__ void cvt_bf16(const float* __restrict__ in,
                         unsigned short* __restrict__ out, long n) {
  long i = (long)blockIdx.x * 256 + threadIdx.x;
  if (i < n) out[i] = f2bf(in[i]);
}

// ---------------- Flash MLA attention --------------------------------------
// grid = (S/64, B*NH), block = 128 (4 waves). Wave w owns q-rows [q0+16w, +16).
// Q fragments (18 x v16bf) in registers; K tile 32x576 bf16 double-buffered in
// LDS, filled with GLOBAL_LOAD_ASYNC_TO_LDS_B128 (ASYNCcnt) when available so
// tile kb+1 streams in while WMMAs consume tile kb; 512-wide f32 accumulator
// (32 v8f) in VGPRs; out_absorb (512->128 per head) fused in epilogue.
__global__ __launch_bounds__(128) void mla_attn(
    const unsigned short* __restrict__ qcat,   // [B*NH, S, 576]
    const unsigned short* __restrict__ kcat,   // [B, S, 576]
    const unsigned short* __restrict__ wkv,    // Wkv_up bf16 [16*256, 512]
    float* __restrict__ attn_out,              // [B, S, 2048]
    int S)
{
  __shared__ __align__(16) unsigned short Ks[2][32 * 576];
  __shared__ __align__(16) unsigned short Pls[4 * 16 * 32];
  int z = blockIdx.y;
  int b = z >> 4, h = z & 15;
  int q0blk = blockIdx.x * 64;
  int tid = threadIdx.x, wave = tid >> 5, lane = tid & 31;
  int qrow0 = q0blk + wave * 16;
  unsigned short* P = Pls + wave * 512;

  const unsigned short* kbase = kcat + (size_t)b * S * 576;

  auto issue_tile = [&](int kb, int buf) {
    const unsigned short* src = kbase + (size_t)kb * 32 * 576;
#if HAVE_ASYNC_LDS
#pragma unroll
    for (int i = 0; i < 18; ++i) {
      int e = (i * 128 + tid) * 8;            // 8 bf16 = 16B per lane
      __builtin_amdgcn_global_load_async_to_lds_b128(
          (AS1 v4i*)(src + e), (AS3 v4i*)&Ks[buf][e], 0, 0);
    }
#else
    const uint4* sv = (const uint4*)src;
    uint4* dv = (uint4*)Ks[buf];
#pragma unroll
    for (int i = 0; i < 18; ++i) dv[i * 128 + tid] = sv[i * 128 + tid];
#endif
  };

  // Preload this wave's Q fragments (16 rows x 576) from global bf16.
  v16bf qf[18];
  const unsigned short* qbase = qcat + ((size_t)z * S + qrow0) * 576;
#pragma unroll
  for (int c = 0; c < 18; ++c) qf[c] = load_afrag(qbase + c * 32, 576);

  v8f o[32];
#pragma unroll
  for (int t = 0; t < 32; ++t) o[t] = v8f_zero();
  float mrow[8], lrow[8];
#pragma unroll
  for (int j = 0; j < 8; ++j) { mrow[j] = -1e30f; lrow[j] = 0.f; }

  const float scale = 0.07216878364870323f;   // 1/sqrt(192)
  int mb = (lane >> 4) * 8, nn = lane & 15;
  int kblocks = (q0blk + 63) / 32 + 1;        // causal upper bound

  issue_tile(0, 0);
  for (int kb = 0; kb < kblocks; ++kb) {
    int cur = kb & 1;
#if HAVE_ASYNC_LDS
    wait_asynccnt0();                         // my async writes to Ks[cur] done
#endif
    __syncthreads();                          // all writes visible; prev reads done
    if (kb + 1 < kblocks) issue_tile(kb + 1, cur ^ 1);
    const unsigned short* Kt = Ks[cur];

    // Scores: S[16x32] = Q(16x576) * K^T, two 16x16 WMMA C tiles
    v8f s0 = v8f_zero(), s1 = v8f_zero();
#pragma unroll
    for (int c = 0; c < 18; ++c) {
      v16bf b0 = load_bfrag_kcont(Kt + 0 * 16 * 576 + c * 32, 576);
      v16bf b1 = load_bfrag_kcont(Kt + 1 * 16 * 576 + c * 32, 576);
      s0 = wmma_bf16(qf[c], b0, s0);
      s1 = wmma_bf16(qf[c], b1, s1);
    }

    // Online softmax (rows over VGPR j + half-wave; cols over 16 lanes)
    float corr[8], p0[8], p1[8];
#pragma unroll
    for (int j = 0; j < 8; ++j) {
      int qg = qrow0 + mb + j;
      int kg0 = kb * 32 + nn;
      float a0 = (kg0      <= qg) ? s0[j] * scale : -1e30f;
      float a1 = (kg0 + 16 <= qg) ? s1[j] * scale : -1e30f;
      float mx = fmaxf(a0, a1);
#pragma unroll
      for (int off = 1; off < 16; off <<= 1) mx = fmaxf(mx, __shfl_xor(mx, off, 32));
      float mnew = fmaxf(mrow[j], mx);
      corr[j] = __expf(mrow[j] - mnew);
      p0[j] = __expf(a0 - mnew);
      p1[j] = __expf(a1 - mnew);
      float rs = p0[j] + p1[j];
#pragma unroll
      for (int off = 1; off < 16; off <<= 1) rs += __shfl_xor(rs, off, 32);
      lrow[j] = lrow[j] * corr[j] + rs;
      mrow[j] = mnew;
    }
#pragma unroll
    for (int t = 0; t < 32; ++t)
#pragma unroll
      for (int j = 0; j < 8; ++j) o[t][j] *= corr[j];

    // Re-layout P (C-layout) -> A-fragment via per-wave LDS scratch (DS in-order)
#pragma unroll
    for (int j = 0; j < 8; ++j) {
      P[(mb + j) * 32 + nn]      = f2bf(p0[j]);
      P[(mb + j) * 32 + nn + 16] = f2bf(p1[j]);
    }
    v16bf ap = load_afrag(P, 32);
    // o[16x512] += P(16x32) * ckv_tile(32x512)
#pragma unroll
    for (int t = 0; t < 32; ++t) {
      v16bf bv = load_bfrag_kstride(Kt + 64 + t * 16, 576);
      o[t] = wmma_bf16(ap, bv, o[t]);
    }
  }

  // Epilogue: normalize and fuse out_absorb[h] (128x512): final[16,128]
  float invl[8];
#pragma unroll
  for (int j = 0; j < 8; ++j) invl[j] = 1.0f / lrow[j];

  v8f fc[8];
#pragma unroll
  for (int t = 0; t < 8; ++t) fc[t] = v8f_zero();
  const unsigned short* wb = wkv + (size_t)(h * 256 + 128) * 512;
#pragma unroll
  for (int cc = 0; cc < 16; ++cc) {           // 512 / 32 contraction chunks
#pragma unroll
    for (int j = 0; j < 8; ++j) {
      P[(mb + j) * 32 + nn]      = f2bf(o[cc * 2][j] * invl[j]);
      P[(mb + j) * 32 + nn + 16] = f2bf(o[cc * 2 + 1][j] * invl[j]);
    }
    v16bf ao = load_afrag(P, 32);
#pragma unroll
    for (int nt = 0; nt < 8; ++nt) {
      v16bf bw = load_bfrag_kcont(wb + (size_t)(nt * 16) * 512 + cc * 32, 512);
      fc[nt] = wmma_bf16(ao, bw, fc[nt]);
    }
  }
#pragma unroll
  for (int nt = 0; nt < 8; ++nt)
#pragma unroll
    for (int j = 0; j < 8; ++j)
      attn_out[((size_t)b * S + qrow0 + mb + j) * 2048 + h * 128 + nt * 16 + nn] =
          fc[nt][j];
}

// ---------------- Host orchestration ---------------------------------------
extern "C" void kernel_launch(void* const* d_in, const int* in_sizes, int n_in,
                              void* d_out, int out_size, void* d_ws, size_t ws_size,
                              hipStream_t stream) {
  const float* hs       = (const float*)d_in[0];
  const int*   pos      = (const int*)  d_in[1];
  // d_in[2] attention_mask: exact tril -> causal mask applied analytically
  const float* Wq_down  = (const float*)d_in[3];
  const float* q_ln_w   = (const float*)d_in[4];
  const float* Wq_up    = (const float*)d_in[5];
  const float* Wkv_down = (const float*)d_in[6];
  const float* Wkv_up   = (const float*)d_in[7];
  const float* Wo       = (const float*)d_in[8];
  float* out = (float*)d_out;
  (void)in_sizes; (void)n_in; (void)out_size; (void)ws_size;

  const int S = 2048;
  const size_t ROWS = 4096;                   // B*S

  size_t off = 0;
  auto take = [&](size_t bytes) {
    size_t o = off; off += (bytes + 255) & ~(size_t)255; return o;
  };
  char* ws = (char*)d_ws;
  float* ql            = (float*)(ws + take(ROWS * 1536 * 4));
  float* qbuf          = (float*)(ws + take(ROWS * 3072 * 4));
  float* ckv           = (float*)(ws + take(ROWS * 576 * 4));
  unsigned short* kcat = (unsigned short*)(ws + take(ROWS * 576 * 2));
  unsigned short* qcat = (unsigned short*)(ws + take((size_t)32 * S * 576 * 2));
  unsigned short* wkvb = (unsigned short*)(ws + take((size_t)8192 * 512 * 2));
  float* aout          = (float*)(ws + take(ROWS * 2048 * 4));

  dim3 blk(128);
  // 1. ql = hs @ Wq_down^T            [4096,1536] k=2048
  gemm_nt<false><<<dim3(24, 32, 1), blk, 0, stream>>>(hs, Wq_down, ql,
      2048, 2048, 2048, 1, 1536, 0, 0, 0, 0, 1);
  // 2. RMSNorm in place
  rmsnorm1536<<<dim3((unsigned)ROWS), dim3(256), 0, stream>>>(ql, q_ln_w);
  // 3. q = ql @ Wq_up^T               [4096,3072] k=1536
  gemm_nt<false><<<dim3(48, 32, 1), blk, 0, stream>>>(ql, Wq_up, qbuf,
      1536, 1536, 1536, 1, 3072, 0, 0, 0, 0, 1);
  // 4. ckv = hs @ Wkv_down^T          [4096,576] k=2048
  gemm_nt<false><<<dim3(9, 32, 1), blk, 0, stream>>>(hs, Wkv_down, ckv,
      2048, 2048, 2048, 1, 576, 0, 0, 0, 0, 1);
  // 5. Wkv_up -> bf16 (for fused out_absorb in attention)
  cvt_bf16<<<dim3(16384), dim3(256), 0, stream>>>(Wkv_up, wkvb, (long)8192 * 512);
  // 6. kcat = [rope(k_rope) ; ckv] bf16
  build_kcat<<<dim3((unsigned)ROWS), dim3(576), 0, stream>>>(ckv, pos, kcat);
  // 7. qcat[:, 0:64] = rope(q_rope)
  rope_q<<<dim3(16384), dim3(256), 0, stream>>>(qbuf, pos, qcat, S);
  // 8. qcat[:, 64:576] = q_nope @ q_absorb[h]   (batched over z = b*16+h)
  gemm_nt<true><<<dim3(8, 16, 32), blk, 0, stream>>>(qbuf, Wkv_up, qcat + 64,
      128, 3072, 1, 512, 576,
      (long)S * 3072 /*per b*/, 192 /*per h*/, (long)256 * 512 /*per h*/,
      (long)S * 576 /*C per z*/, 16);
  // 9. Flash MLA attention with fused out_absorb -> aout [B,S,2048]
  mla_attn<<<dim3(32, 32), blk, 0, stream>>>(qcat, kcat, wkvb, aout, S);
  // 10. out = aout @ Wo^T             [4096,2048] k=2048
  gemm_nt<false><<<dim3(32, 32, 1), blk, 0, stream>>>(aout, Wo, out,
      2048, 2048, 2048, 1, 2048, 0, 0, 0, 0, 1);
}
